// BiRNN_67946382623439
// MI455X (gfx1250) — matrix operand order repaired and checked by reading, not compiled, not run on hardware
//
#include <hip/hip_runtime.h>
#include <hip/hip_bf16.h>
#include <math.h>

// Problem dims (match reference)
constexpr int B_ = 64;        // batch
constexpr int S_ = 512;       // seq len
constexpr int E_ = 512;       // embedding dim
constexpr int H_ = 1024;      // hidden dim
constexpr int NWGD = 16;      // workgroups per direction in recurrent kernel
constexpr int HC = H_ / NWGD; // 64 columns per WG
constexpr int LDSP = H_ + 8;  // padded LDS row stride (elems): 2064B -> 4-bank stagger/row

typedef __attribute__((ext_vector_type(16))) __bf16 v16bf;
typedef __attribute__((ext_vector_type(8)))  __bf16 v8bf;
typedef __attribute__((ext_vector_type(8)))  float  v8f;

union Frag16 { v16bf v; v8bf h[2]; };

__device__ __forceinline__ v8f wmma_bf16(v16bf a, v16bf b, v8f c) {
  // D = A(16x32) * B(32x16) + C, f32 accumulate
  return __builtin_amdgcn_wmma_f32_16x16x32_bf16(false, a, false, b, (short)0, c,
                                                 false, false);
}

// A fragment (16x32 bf16) from row-major bf16 [rows][ld] in GLOBAL memory.
__device__ __forceinline__ v16bf load_a_frag_g(const __bf16* A, int ld, int mt,
                                               int k0, int lane) {
  int m  = mt * 16 + (lane & 15);
  int kb = k0 + ((lane >> 4) << 3);
  const __bf16* p = A + (size_t)m * ld + kb;
  Frag16 f;
  f.h[0] = *(const v8bf*)(p);
  f.h[1] = *(const v8bf*)(p + 16);
  return f.v;
}

// A fragment from an LDS-staged [64][LDSP] buffer (generic ptr into __shared__).
__device__ __forceinline__ v16bf load_a_frag_lds(const __bf16* A, int mt,
                                                 int k0, int lane) {
  int m  = mt * 16 + (lane & 15);
  int kb = k0 + ((lane >> 4) << 3);
  const __bf16* p = A + m * LDSP + kb;
  Frag16 f;
  f.h[0] = *(const v8bf*)(p);        // ds_load_b128
  f.h[1] = *(const v8bf*)(p + 16);
  return f.v;
}

// B fragment (32x16 bf16) from pre-transposed weights WT[N][K] (row-major).
__device__ __forceinline__ v16bf load_b_frag(const __bf16* WT, int ldk, int n0,
                                             int k0, int lane) {
  int n  = n0 + (lane & 15);
  int kb = k0 + ((lane >> 4) << 3);
  const __bf16* p = WT + (size_t)n * ldk + kb;
  Frag16 f;
  f.h[0] = *(const v8bf*)(p);
  f.h[1] = *(const v8bf*)(p + 16);
  return f.v;
}

// ---- CDNA5 async global->LDS copy (ASYNCcnt path), inline asm: portable across toolchains.
// ldsaddr: wave-relative LDS byte address (low 32 bits of a generic __shared__ pointer).
__device__ __forceinline__ void async_b128_to_lds(unsigned ldsaddr, const void* gaddr) {
  asm volatile("global_load_async_to_lds_b128 %0, %1, off"
               :: "v"(ldsaddr), "v"(gaddr) : "memory");
}
// Copy one [64][1024] bf16 matrix (128KB) from global into a padded LDS buffer.
// 8192 16B chunks over 256 threads -> 32 async issues per thread (in-order batch).
__device__ __forceinline__ void async_h_to_lds(const __bf16* src, __bf16* ldsbase,
                                               int tid) {
  unsigned lbase = (unsigned)(uintptr_t)ldsbase;
#pragma unroll 4
  for (int c = tid; c < (B_ * H_ * 2) / 16; c += 256) {
    int row = c >> 7;        // 128 x 16B chunks per 2048B row
    int col = c & 127;
    unsigned laddr = lbase + (unsigned)(row * (LDSP * 2) + col * 16);
    const char* g = (const char*)src + ((size_t)row * (H_ * 2) + (size_t)col * 16);
    async_b128_to_lds(laddr, g);
  }
}
__device__ __forceinline__ void wait_async_le(int) {}
#define WAIT_ASYNC_32() asm volatile("s_wait_asynccnt 32" ::: "memory")
#define WAIT_ASYNC_0()  asm volatile("s_wait_asynccnt 0"  ::: "memory")

// ---------------- prep kernels ----------------

__global__ __launch_bounds__(256) void transpose_to_bf16(const float* __restrict__ in,
                                                         __bf16* __restrict__ out,
                                                         int K, int N) {
  size_t idx = (size_t)blockIdx.x * 256 + threadIdx.x;
  if (idx >= (size_t)K * N) return;
  int k = (int)(idx / N);
  int n = (int)(idx % N);
  out[(size_t)n * K + k] = (__bf16)in[idx];
}

__global__ __launch_bounds__(256) void bias_sum(const float* __restrict__ a,
                                                const float* __restrict__ b,
                                                float* __restrict__ out, int n) {
  int i = blockIdx.x * 256 + threadIdx.x;
  if (i < n) out[i] = a[i] + b[i];
}

__global__ void init_bars(unsigned* bars) {
  if (threadIdx.x < 8) bars[threadIdx.x] = 0u;
}

// ---------------- phase A: Xproj[dir][s][b][:] = emb[x[b][s]] @ W0T + bias0 ----------------

__global__ __launch_bounds__(256) void embed_proj(
    const int* __restrict__ x,          // [B][S]
    const float* __restrict__ emb,      // [V][E]
    const __bf16* __restrict__ W0T,     // [2][H][E] bf16 (transposed)
    const float* __restrict__ bias0,    // [2][H]  (bih0 + bhh0)
    __bf16* __restrict__ Xproj)         // [2][S][B][H] bf16
{
  __shared__ __bf16 ldsA[B_][E_ + 8];   // 64 x 520 bf16; +8 pad kills bank conflicts

  int s   = blockIdx.x;
  int ct  = blockIdx.y;                 // column tile (16 of them, 64 cols each)
  int dir = blockIdx.z;
  int nbase = ct * 64;
  int tid = threadIdx.x;

  // Stage gathered embedding rows (f32 -> bf16) into LDS: 4 threads per row.
  {
    int row = tid >> 2;
    int q   = tid & 3;
    int tok = x[row * S_ + s];
    const float* er = emb + (size_t)tok * E_ + q * 128;
    __bf16* dst = &ldsA[row][q * 128];
#pragma unroll 4
    for (int i = 0; i < 128; i += 4) {
      float4 f = *(const float4*)(er + i);
      dst[i]     = (__bf16)f.x;
      dst[i + 1] = (__bf16)f.y;
      dst[i + 2] = (__bf16)f.z;
      dst[i + 3] = (__bf16)f.w;
    }
  }
  __syncthreads();

  int lane = tid & 31, w = tid >> 5;
  int mt = w & 3;                 // 4 row tiles of 16
  int nt = (w >> 2) * 32;         // 2 waves cover the 64-col slab, 2 tiles each

  const __bf16* WT = W0T + (size_t)dir * H_ * E_;
  const float*  bi = bias0 + dir * H_;

  v8f acc0 = {}, acc1 = {};
  for (int k0 = 0; k0 < E_; k0 += 32) {
    int m  = mt * 16 + (lane & 15);
    int kb = k0 + ((lane >> 4) << 3);
    Frag16 f;
    f.h[0] = *(const v8bf*)&ldsA[m][kb];       // ds_load_b128
    f.h[1] = *(const v8bf*)&ldsA[m][kb + 16];
    v16bf a = f.v;
    acc0 = wmma_bf16(a, load_b_frag(WT, E_, nbase + nt,      k0, lane), acc0);
    acc1 = wmma_bf16(a, load_b_frag(WT, E_, nbase + nt + 16, k0, lane), acc1);
  }

  __bf16* out = Xproj + ((size_t)dir * S_ + s) * B_ * H_;
  int n0 = nbase + nt + (lane & 15);
  int mb = mt * 16 + ((lane >> 4) << 3);
#pragma unroll
  for (int r = 0; r < 8; ++r) {
    out[(size_t)(mb + r) * H_ + n0]      = (__bf16)(acc0[r] + bi[n0]);
    out[(size_t)(mb + r) * H_ + n0 + 16] = (__bf16)(acc1[r] + bi[n0 + 16]);
  }
}

// ---------------- phase B: sequential recurrence (persistent, 16 WGs / direction) ----------------

__device__ __forceinline__ void dir_barrier(unsigned* bar, unsigned nwg) {
  __threadfence();        // flush this thread's global writes (device scope)
  __syncthreads();
  if (threadIdx.x == 0) {
    unsigned gen  = __hip_atomic_load(bar + 1, __ATOMIC_RELAXED, __HIP_MEMORY_SCOPE_AGENT);
    unsigned prev = __hip_atomic_fetch_add(bar, 1u, __ATOMIC_ACQ_REL, __HIP_MEMORY_SCOPE_AGENT);
    if (prev == nwg - 1) {
      __hip_atomic_store(bar, 0u, __ATOMIC_RELAXED, __HIP_MEMORY_SCOPE_AGENT);
      __hip_atomic_fetch_add(bar + 1, 1u, __ATOMIC_RELEASE, __HIP_MEMORY_SCOPE_AGENT);
    } else {
      while (__hip_atomic_load(bar + 1, __ATOMIC_ACQUIRE, __HIP_MEMORY_SCOPE_AGENT) == gen)
        __builtin_amdgcn_s_sleep(1);
    }
  }
  __syncthreads();
  __threadfence();        // acquire side: invalidate stale cached h
}

__device__ __forceinline__ void store_tanh_add(__bf16* dst, const __bf16* addrow,
                                               v8f acc, int n0, int mt, int lane) {
  int n  = n0 + (lane & 15);
  int mb = mt * 16 + ((lane >> 4) << 3);
#pragma unroll
  for (int r = 0; r < 8; ++r) {
    float v = acc[r] + (float)addrow[(size_t)(mb + r) * H_ + n];
    dst[(size_t)(mb + r) * H_ + n] = (__bf16)tanhf(v);
  }
}

__device__ __forceinline__ void store_tanh_bias(__bf16* dst, const float* bias,
                                                v8f acc, int n0, int mt, int lane) {
  int n  = n0 + (lane & 15);
  float bv = bias[n];
  int mb = mt * 16 + ((lane >> 4) << 3);
#pragma unroll
  for (int r = 0; r < 8; ++r)
    dst[(size_t)(mb + r) * H_ + n] = (__bf16)tanhf(acc[r] + bv);
}

__global__ __launch_bounds__(256) void rnn_recurrent(
    const __bf16* __restrict__ Xproj,  // [2][S][B][H]
    const __bf16* __restrict__ Wrec,   // [2][3][H][H]: Whh0T, Wih1T, Whh1T
    const float*  __restrict__ bias1,  // [2][H] (bih1 + bhh1)
    __bf16* h0,                        // [2][2][B][H] double-buffered
    __bf16* h1,                        // [2][2][B][H]
    unsigned* bars)                    // [2][2] (arrival, generation) per direction
{
  // Async-staged A operands: ldsA0 <- h0_prev (prefetched during previous layer-1),
  // ldsA1 <- h1_prev (prefetched at step start). 2 x 132KB = 264KB (<320KB/WGP).
  __shared__ __bf16 ldsA0[B_][LDSP];
  __shared__ __bf16 ldsA1[B_][LDSP];

  int dir = blockIdx.x >> 4;
  int wg  = blockIdx.x & 15;
  int nbase = wg * HC;
  int tid  = threadIdx.x;
  int lane = tid & 31;
  int w    = tid >> 5;
  int mt = w & 3;
  int nt = (w >> 2) * 32;

  const __bf16* Whh0T = Wrec + ((size_t)dir * 3 + 0) * H_ * H_;
  const __bf16* Wih1T = Wrec + ((size_t)dir * 3 + 1) * H_ * H_;
  const __bf16* Whh1T = Wrec + ((size_t)dir * 3 + 2) * H_ * H_;
  const __bf16* Xp = Xproj + (size_t)dir * S_ * B_ * H_;
  __bf16* h0d = h0 + (size_t)dir * 2 * B_ * H_;
  __bf16* h1d = h1 + (size_t)dir * 2 * B_ * H_;
  unsigned* bar = bars + dir * 2;
  const float* b1 = bias1 + dir * H_;

  // zero-init buffer 0 (this WG's column slice), then sync the direction
  for (int i = tid; i < B_ * HC; i += 256) {
    int b = i / HC, c = i % HC;
    h0d[(size_t)b * H_ + nbase + c] = (__bf16)0.f;
    h1d[(size_t)b * H_ + nbase + c] = (__bf16)0.f;
  }
  dir_barrier(bar, NWGD);

  // Preload h0_prev(t=0) = zeros buffer into ldsA0 (batch of 32 async issues/wave).
  async_h_to_lds(h0d, &ldsA0[0][0], tid);

  for (int t = 0; t < S_; ++t) {
    int ts = dir ? (S_ - 1 - t) : t;   // backward direction walks the sequence reversed
    int pb = t & 1, cb = pb ^ 1;
    const __bf16* h0p = h0d + (size_t)pb * B_ * H_;
    __bf16*       h0c = h0d + (size_t)cb * B_ * H_;
    const __bf16* h1p = h1d + (size_t)pb * B_ * H_;
    __bf16*       h1c = h1d + (size_t)cb * B_ * H_;
    const __bf16* xrow = Xp + (size_t)ts * B_ * H_;

    // Kick off h1_prev -> ldsA1; consumed only after the mid-step barrier.
    async_h_to_lds(h1p, &ldsA1[0][0], tid);

    // Drain the OLDER batch (h0_prev -> ldsA0); async loads retire in order,
    // so waiting for <=32 outstanding leaves the h1p batch in flight.
    WAIT_ASYNC_32();
    __syncthreads();

    // layer 0: h0c = tanh(Xproj[ts] + h0p @ Whh0T)   (A from LDS)
    v8f acc0 = {}, acc1 = {};
    for (int k0 = 0; k0 < H_; k0 += 32) {
      v16bf a = load_a_frag_lds(&ldsA0[0][0], mt, k0, lane);
      acc0 = wmma_bf16(a, load_b_frag(Whh0T, H_, nbase + nt,      k0, lane), acc0);
      acc1 = wmma_bf16(a, load_b_frag(Whh0T, H_, nbase + nt + 16, k0, lane), acc1);
    }
    store_tanh_add(h0c, xrow, acc0, nbase + nt,      mt, lane);
    store_tanh_add(h0c, xrow, acc1, nbase + nt + 16, mt, lane);
    dir_barrier(bar, NWGD);   // full h0c visible to every WG of this direction

    // Prefetch next step's h0_prev (= h0c) into ldsA0; overlaps all of layer 1.
    if (t + 1 < S_) async_h_to_lds(h0c, &ldsA0[0][0], tid);

    // layer 1 part A: acc += h0c @ Wih1T   (A straight from global; h0c is L2-hot)
    acc0 = {}; acc1 = {};
    for (int k0 = 0; k0 < H_; k0 += 32) {
      v16bf a = load_a_frag_g(h0c, H_, mt, k0, lane);
      acc0 = wmma_bf16(a, load_b_frag(Wih1T, H_, nbase + nt,      k0, lane), acc0);
      acc1 = wmma_bf16(a, load_b_frag(Wih1T, H_, nbase + nt + 16, k0, lane), acc1);
    }

    // Drain the h1p batch (older of the outstanding batches), then consume ldsA1.
    if (t + 1 < S_) { WAIT_ASYNC_32(); } else { WAIT_ASYNC_0(); }
    __syncthreads();

    // layer 1 part B: acc += h1p @ Whh1T   (A from LDS)
    for (int k0 = 0; k0 < H_; k0 += 32) {
      v16bf a = load_a_frag_lds(&ldsA1[0][0], mt, k0, lane);
      acc0 = wmma_bf16(a, load_b_frag(Whh1T, H_, nbase + nt,      k0, lane), acc0);
      acc1 = wmma_bf16(a, load_b_frag(Whh1T, H_, nbase + nt + 16, k0, lane), acc1);
    }
    store_tanh_bias(h1c, b1, acc0, nbase + nt,      mt, lane);
    store_tanh_bias(h1c, b1, acc1, nbase + nt + 16, mt, lane);
    dir_barrier(bar, NWGD);   // h1c visible; h0p/h1p buffers free for reuse next step
  }
}

// ---------------- phase C: classifier head (exact f32, tiny) ----------------

__global__ __launch_bounds__(64) void classifier(
    const __bf16* __restrict__ h1F,   // [B][H] final fwd hidden (buffer 0, S even)
    const __bf16* __restrict__ h1B,   // [B][H] final bwd hidden
    const float* __restrict__ fc1_w,  // [2H][64]
    const float* __restrict__ fc1_b,  // [64]
    const float* __restrict__ fc2_w,  // [64][2]
    const float* __restrict__ fc2_b,  // [2]
    float* __restrict__ out)          // [B][2]
{
  int b = threadIdx.x;
  float mid[64];
#pragma unroll
  for (int j = 0; j < 64; ++j) mid[j] = fc1_b[j];
  for (int k = 0; k < H_; ++k) {
    float hf = (float)h1F[(size_t)b * H_ + k];
#pragma unroll 8
    for (int j = 0; j < 64; ++j) mid[j] += hf * fc1_w[(size_t)k * 64 + j];
  }
  for (int k = 0; k < H_; ++k) {
    float hb = (float)h1B[(size_t)b * H_ + k];
#pragma unroll 8
    for (int j = 0; j < 64; ++j) mid[j] += hb * fc1_w[(size_t)(H_ + k) * 64 + j];
  }
  float o0 = fc2_b[0], o1 = fc2_b[1];
#pragma unroll
  for (int j = 0; j < 64; ++j) {
    o0 += mid[j] * fc2_w[j * 2 + 0];
    o1 += mid[j] * fc2_w[j * 2 + 1];
  }
  out[b * 2 + 0] = o0;
  out[b * 2 + 1] = o1;
}

// ---------------- host ----------------

extern "C" void kernel_launch(void* const* d_in, const int* in_sizes, int n_in,
                              void* d_out, int out_size, void* d_ws, size_t ws_size,
                              hipStream_t stream) {
  (void)in_sizes; (void)n_in; (void)out_size; (void)ws_size;

  const int*   x       = (const int*)d_in[0];
  const float* emb     = (const float*)d_in[1];
  const float* fw0_wih = (const float*)d_in[2];
  const float* fw0_bih = (const float*)d_in[3];
  const float* fw0_whh = (const float*)d_in[4];
  const float* fw0_bhh = (const float*)d_in[5];
  const float* fw1_wih = (const float*)d_in[6];
  const float* fw1_bih = (const float*)d_in[7];
  const float* fw1_whh = (const float*)d_in[8];
  const float* fw1_bhh = (const float*)d_in[9];
  const float* bw0_wih = (const float*)d_in[10];
  const float* bw0_bih = (const float*)d_in[11];
  const float* bw0_whh = (const float*)d_in[12];
  const float* bw0_bhh = (const float*)d_in[13];
  const float* bw1_wih = (const float*)d_in[14];
  const float* bw1_bih = (const float*)d_in[15];
  const float* bw1_whh = (const float*)d_in[16];
  const float* bw1_bhh = (const float*)d_in[17];
  const float* fc1_w   = (const float*)d_in[18];
  const float* fc1_b   = (const float*)d_in[19];
  const float* fc2_w   = (const float*)d_in[20];
  const float* fc2_b   = (const float*)d_in[21];

  char* ws = (char*)d_ws;
  size_t off = 0;
  __bf16* Xproj = (__bf16*)(ws + off); off += (size_t)2 * S_ * B_ * H_ * 2;     // 128 MB
  __bf16* Wrec  = (__bf16*)(ws + off); off += (size_t)2 * 3 * H_ * H_ * 2;      // 12 MB
  __bf16* W0T   = (__bf16*)(ws + off); off += (size_t)2 * H_ * E_ * 2;          // 2 MB
  float*  bias0 = (float*)(ws + off);  off += (size_t)2 * H_ * 4;
  float*  bias1 = (float*)(ws + off);  off += (size_t)2 * H_ * 4;
  __bf16* h0    = (__bf16*)(ws + off); off += (size_t)2 * 2 * B_ * H_ * 2;
  __bf16* h1    = (__bf16*)(ws + off); off += (size_t)2 * 2 * B_ * H_ * 2;
  unsigned* bars = (unsigned*)(ws + off); off += 256;

  // Biases: layer0 biases fold into Xproj; layer1 biases added each step.
  int bgrid = (H_ + 255) / 256;
  bias_sum<<<bgrid, 256, 0, stream>>>(fw0_bih, fw0_bhh, bias0,       H_);
  bias_sum<<<bgrid, 256, 0, stream>>>(bw0_bih, bw0_bhh, bias0 + H_,  H_);
  bias_sum<<<bgrid, 256, 0, stream>>>(fw1_bih, fw1_bhh, bias1,       H_);
  bias_sum<<<bgrid, 256, 0, stream>>>(bw1_bih, bw1_bhh, bias1 + H_,  H_);

  // Weights: f32 [K][N] -> bf16 [N][K] so B-fragments are contiguous along K.
  int tg_eh = (E_ * H_ + 255) / 256;
  int tg_hh = (H_ * H_ + 255) / 256;
  transpose_to_bf16<<<tg_eh, 256, 0, stream>>>(fw0_wih, W0T,              E_, H_);
  transpose_to_bf16<<<tg_eh, 256, 0, stream>>>(bw0_wih, W0T + (size_t)H_ * E_, E_, H_);
  transpose_to_bf16<<<tg_hh, 256, 0, stream>>>(fw0_whh, Wrec + (size_t)0 * H_ * H_, H_, H_);
  transpose_to_bf16<<<tg_hh, 256, 0, stream>>>(fw1_wih, Wrec + (size_t)1 * H_ * H_, H_, H_);
  transpose_to_bf16<<<tg_hh, 256, 0, stream>>>(fw1_whh, Wrec + (size_t)2 * H_ * H_, H_, H_);
  transpose_to_bf16<<<tg_hh, 256, 0, stream>>>(bw0_whh, Wrec + (size_t)3 * H_ * H_, H_, H_);
  transpose_to_bf16<<<tg_hh, 256, 0, stream>>>(bw1_wih, Wrec + (size_t)4 * H_ * H_, H_, H_);
  transpose_to_bf16<<<tg_hh, 256, 0, stream>>>(bw1_whh, Wrec + (size_t)5 * H_ * H_, H_, H_);

  init_bars<<<1, 32, 0, stream>>>(bars);

  // Phase A: hoisted input projection for both directions (fully parallel WMMA GEMM).
  embed_proj<<<dim3(S_, 16, 2), 256, 0, stream>>>(x, emb, W0T, bias0, Xproj);

  // Phase B: persistent recurrence; 16 WGs per direction, software barrier per half-step,
  // async global->LDS prefetch of the A operands overlapping compute.
  rnn_recurrent<<<32, 256, 0, stream>>>(Xproj, Wrec, bias1, h0, h1, bars);

  // Phase C: final hidden lives in buffer 0 (S even). bwd dir base = h1 + 2*B*H.
  classifier<<<1, 64, 0, stream>>>(h1, h1 + (size_t)2 * B_ * H_,
                                   fc1_w, fc1_b, fc2_w, fc2_b, (float*)d_out);
}
